// MultiHeadAttention_51960514347239
// MI455X (gfx1250) — matrix operand order repaired and checked by reading, not compiled, not run on hardware
//
#include <hip/hip_runtime.h>
#include <hip/hip_bf16.h>

#define DEV __device__ __forceinline__

typedef __attribute__((ext_vector_type(16))) __bf16          v16bf;
typedef __attribute__((ext_vector_type(8)))  float           v8f;
typedef __attribute__((ext_vector_type(8)))  unsigned short  v8us;
typedef __attribute__((ext_vector_type(16))) unsigned short  v16us;

union BFP { v16us u; v16bf b; };

// Problem constants (fixed by the reference).
#define Bn  4
#define Ln  2048
#define Dn  1024
#define Hn  8
#define DHn 128
#define Mn  (Bn * Ln)          // 8192 rows
#define NEGV (-4294967295.0f)  // -(2^32 - 1)

// ---------- bf16 helpers ----------
DEV unsigned short f2bf(float f) {
  union { float f; unsigned u; } x; x.f = f;
  unsigned u = x.u;
  u += 0x7fffu + ((u >> 16) & 1u);   // round-to-nearest-even
  return (unsigned short)(u >> 16);
}

DEV v8us cvt8(const float* p) {
  v8us r;
#pragma unroll
  for (int i = 0; i < 8; ++i) r[i] = f2bf(p[i]);
  return r;
}
DEV v8us cvt8(const unsigned short* p) {   // already bf16 bits
  return *(const v8us*)p;
}

DEV v16bf combine16(v8us lo, v8us hi) {
  BFP u;
  u.u = __builtin_shufflevector(lo, hi, 0,1,2,3,4,5,6,7,8,9,10,11,12,13,14,15);
  return u.b;
}

// A-matrix 16x32 bf16 fragment (ISA 7.12.2): lane m holds K={kh*8..+7, 16+kh*8..+7}
template <typename T>
DEV v16bf load_afrag(const T* X, long ld, long row, int k0, int lane) {
  const int kh = (lane >> 4) & 1;
  const T* p = X + row * ld + k0 + kh * 8;
  return combine16(cvt8(p), cvt8(p + 16));
}

// B-matrix 32x16 bf16 fragment: lane n holds K={kh*8..+7, 16+kh*8..+7} (same as A).
// "row" is the N index of a row-major (N,K) source (W, K-rows, or V^T rows).
template <typename T>
DEV v16bf load_bfrag(const T* Wm, long ld, long row, long k0, int lane) {
  const int kh = (lane >> 4) & 1;
  const T* p = Wm + row * ld + k0 + kh * 8;
  return combine16(cvt8(p), cvt8(p + 16));
}

DEV v8f wmma_bf16(v16bf a, v16bf b, v8f c) {
  return __builtin_amdgcn_wmma_f32_16x16x32_bf16(
      /*neg_a=*/false, a, /*neg_b=*/false, b,
      /*c_mod=*/(short)0, c, /*reuse_a=*/false, /*reuse_b=*/false);
}

DEV void stout(float* p, float v)          { *p = v; }
DEV void stout(unsigned short* p, float v) { *p = f2bf(v); }

// ---------- elementwise fp32 -> bf16 convert (one pass, removes cvt from GEMMs) ----
__global__ __launch_bounds__(256)
void cvt_f32_bf16(const float* __restrict__ src, unsigned short* __restrict__ dst,
                  long n) {
  const long i = ((long)blockIdx.x * 256 + threadIdx.x) * 8;
  if (i + 8 <= n) {
    const float4 a = *(const float4*)(src + i);
    const float4 b = *(const float4*)(src + i + 4);
    v8us r;
    r[0] = f2bf(a.x); r[1] = f2bf(a.y); r[2] = f2bf(a.z); r[3] = f2bf(a.w);
    r[4] = f2bf(b.x); r[5] = f2bf(b.y); r[6] = f2bf(b.z); r[7] = f2bf(b.w);
    *(v8us*)(dst + i) = r;
  }
}

// ---------- GEMM: out[m,n] = sum_k X[m,k] * W[n,k]  (x @ W^T) ----------
// 128 threads = 4 waves, each wave a 64x64 tile -> 128x128 block tile.
// Strided output so the same kernel writes row-major (Q,K,final) or transposed (V^T).
template <typename TA, typename TB, typename TOUT>
__global__ __launch_bounds__(128)
void gemm_xwT(const TA* __restrict__ X, const TB* __restrict__ Wm,
              TOUT* __restrict__ out, int Kdim, long sm, long sn) {
  const int lane = threadIdx.x & 31;
  const int wv   = threadIdx.x >> 5;
  const int lm   = lane & 15;
  const int rb   = (lane >> 4) * 8;
  const long m0 = (long)blockIdx.y * 128 + (wv >> 1) * 64;
  const long n0 = (long)blockIdx.x * 128 + (wv & 1) * 64;

  const v8f zero8 = {0.f,0.f,0.f,0.f,0.f,0.f,0.f,0.f};
  v8f acc[4][4];
#pragma unroll
  for (int i = 0; i < 4; ++i)
#pragma unroll
    for (int j = 0; j < 4; ++j) acc[i][j] = zero8;

  for (int k0 = 0; k0 < Kdim; k0 += 32) {
    v16bf af[4], bf[4];
#pragma unroll
    for (int i = 0; i < 4; ++i)
      af[i] = load_afrag(X, (long)Kdim, m0 + i * 16 + lm, k0, lane);
#pragma unroll
    for (int j = 0; j < 4; ++j)
      bf[j] = load_bfrag(Wm, (long)Kdim, n0 + j * 16 + lm, (long)k0, lane);
#pragma unroll
    for (int i = 0; i < 4; ++i)
#pragma unroll
      for (int j = 0; j < 4; ++j)
        acc[i][j] = wmma_bf16(af[i], bf[j], acc[i][j]);
  }

#pragma unroll
  for (int i = 0; i < 4; ++i)
#pragma unroll
    for (int j = 0; j < 4; ++j)
#pragma unroll
      for (int v = 0; v < 8; ++v) {
        const long m = m0 + i * 16 + rb + v;
        const long n = n0 + j * 16 + lm;
        stout(out + m * sm + n * sn, acc[i][j][v]);
      }
}

// ---------- Fused flash attention ----------
// Q,K: (B*L, D) bf16 row-major (head h at cols h*128). Vt: (D, B*L) bf16.
// O: (B*L, D) bf16. One wave = 32 query rows (two 16-row A-tiles sharing K/V frags);
// 4 waves/block -> 128 queries/block; grid (L/128, B, H).
__global__ __launch_bounds__(128)
void attn_fused(const unsigned short* __restrict__ Qb,
                const unsigned short* __restrict__ Kb,
                const unsigned short* __restrict__ Vt,
                const int* __restrict__ seqlen,
                unsigned short* __restrict__ Ob) {
  const int lane = threadIdx.x & 31;
  const int wv   = threadIdx.x >> 5;
  const int lm   = lane & 15;
  const int rb   = (lane >> 4) * 8;
  const int kh   = (lane >> 4) & 1;
  const int b = blockIdx.y, h = blockIdx.z;
  const int q0 = blockIdx.x * 128 + wv * 32;
  const int slen = seqlen[b];

  __shared__ __align__(16) float          sS[4][32][33];  // padded scores
  __shared__ __align__(16) unsigned short sP[4][32][32];  // bf16 probabilities
  __shared__ float sStat[4][32];

  // Q fragments: 2 row-tiles x 4 dh-chunks, kept in registers.
  v16bf qf[2][4];
#pragma unroll
  for (int i = 0; i < 2; ++i)
#pragma unroll
    for (int c = 0; c < 4; ++c)
      qf[i][c] = load_afrag(Qb, (long)Dn, (long)b * Ln + q0 + i * 16 + lm,
                            h * DHn + c * 32, lane);

  const v8f zero8 = {0.f,0.f,0.f,0.f,0.f,0.f,0.f,0.f};
  v8f o[2][8];
#pragma unroll
  for (int i = 0; i < 2; ++i)
#pragma unroll
    for (int t = 0; t < 8; ++t) o[i][t] = zero8;

  float m_i = -3.0e38f, l_i = 0.f;          // lane r owns query row r (0..31)
  const float scale = 0.08838834764831843f; // 1/sqrt(128)

  for (int kb = 0; kb < slen; kb += 32) {
    // --- S = Q K^T: batch all K fragment loads, then 16 WMMAs ---
    v16bf kf[2][4];
#pragma unroll
    for (int t = 0; t < 2; ++t)
#pragma unroll
      for (int c = 0; c < 4; ++c)
        kf[t][c] = load_bfrag(Kb, (long)Dn, (long)b * Ln + kb + t * 16 + lm,
                              (long)(h * DHn + c * 32), lane);
    v8f s[2][2] = {{zero8, zero8}, {zero8, zero8}};
#pragma unroll
    for (int c = 0; c < 4; ++c)
#pragma unroll
      for (int i = 0; i < 2; ++i)
#pragma unroll
        for (int t = 0; t < 2; ++t)
          s[i][t] = wmma_bf16(qf[i][c], kf[t][c], s[i][t]);

    // --- Prefetch V fragments; latency hidden behind the softmax below ---
    v16bf vf[8];
#pragma unroll
    for (int t = 0; t < 8; ++t)
      vf[t] = load_bfrag(Vt, (long)Mn, (long)h * DHn + t * 16 + lm,
                         (long)b * Ln + kb, lane);

    // --- Scale + additive key mask, C-layout -> LDS row-major ---
    {
      const int key0 = kb + lm, key1 = kb + 16 + lm;
#pragma unroll
      for (int i = 0; i < 2; ++i)
#pragma unroll
        for (int v = 0; v < 8; ++v) {
          sS[wv][i * 16 + rb + v][lm]      = (key0 < slen) ? s[i][0][v] * scale : NEGV;
          sS[wv][i * 16 + rb + v][16 + lm] = (key1 < slen) ? s[i][1][v] * scale : NEGV;
        }
    }
    asm volatile("s_wait_dscnt 0x0" ::: "memory");
    __builtin_amdgcn_wave_barrier();

    // --- Online softmax: every lane owns one of the 32 query rows ---
    {
      float mx = m_i;
#pragma unroll
      for (int c = 0; c < 32; ++c) mx = fmaxf(mx, sS[wv][lane][c]);
      const float corr = __expf(m_i - mx);
      float sum = 0.f;
#pragma unroll
      for (int c = 0; c < 32; ++c) {
        const float p = __expf(sS[wv][lane][c] - mx);
        sum += p;
        sP[wv][lane][c] = f2bf(p);
      }
      l_i = l_i * corr + sum;
      m_i = mx;
      sStat[wv][lane] = corr;
    }
    asm volatile("s_wait_dscnt 0x0" ::: "memory");
    __builtin_amdgcn_wave_barrier();

    // --- Rescale O accumulators ---
    float corr8[2][8];
#pragma unroll
    for (int i = 0; i < 2; ++i)
#pragma unroll
      for (int v = 0; v < 8; ++v) corr8[i][v] = sStat[wv][i * 16 + rb + v];
#pragma unroll
    for (int i = 0; i < 2; ++i)
#pragma unroll
      for (int t = 0; t < 8; ++t)
#pragma unroll
        for (int v = 0; v < 8; ++v) o[i][t][v] *= corr8[i][v];

    // --- P A-fragments (16x32 each) from LDS in the 16-bit A layout ---
    v16bf pf[2];
#pragma unroll
    for (int i = 0; i < 2; ++i) {
      const unsigned short* pp = &sP[wv][i * 16 + lm][kh * 8];
      v8us lo = *(const v8us*)pp;
      v8us hi = *(const v8us*)(pp + 16);
      pf[i] = combine16(lo, hi);
    }

    // --- O += P @ V : 16 WMMAs reuse the prefetched V fragments ---
#pragma unroll
    for (int t = 0; t < 8; ++t)
#pragma unroll
      for (int i = 0; i < 2; ++i)
        o[i][t] = wmma_bf16(pf[i], vf[t], o[i][t]);
  }

  // 1/l_i and post-softmax query mask (zero rows with q >= seqlen).
  {
    const bool qvalid = (q0 + lane) < slen;
    sStat[wv][lane] = (qvalid && l_i > 0.f) ? (1.f / l_i) : 0.f;
  }
  asm volatile("s_wait_dscnt 0x0" ::: "memory");
  __builtin_amdgcn_wave_barrier();

  float inv8[2][8];
#pragma unroll
  for (int i = 0; i < 2; ++i)
#pragma unroll
    for (int v = 0; v < 8; ++v) inv8[i][v] = sStat[wv][i * 16 + rb + v];
#pragma unroll
  for (int i = 0; i < 2; ++i)
#pragma unroll
    for (int t = 0; t < 8; ++t)
#pragma unroll
      for (int v = 0; v < 8; ++v) {
        const long mrow = (long)b * Ln + q0 + i * 16 + rb + v;
        const int  n    = h * DHn + t * 16 + lm;
        Ob[mrow * Dn + n] = f2bf(o[i][t][v] * inv8[i][v]);
      }
}

// ---------- host-side launch ----------
extern "C" void kernel_launch(void* const* d_in, const int* in_sizes, int n_in,
                              void* d_out, int out_size, void* d_ws, size_t ws_size,
                              hipStream_t stream) {
  (void)in_sizes; (void)n_in; (void)out_size;
  const float* q   = (const float*)d_in[0];
  const float* k   = (const float*)d_in[1];
  const int*   sl  = (const int*)  d_in[2];
  const float* Wq  = (const float*)d_in[3];
  const float* Wk  = (const float*)d_in[4];
  const float* Wv  = (const float*)d_in[5];
  const float* Wo  = (const float*)d_in[6];
  float* out = (float*)d_out;

  const size_t elems = (size_t)Mn * Dn;               // 8M elements
  const size_t welem = (size_t)Dn * Dn;               // 1M per weight
  unsigned short* Qb   = (unsigned short*)d_ws;
  unsigned short* Kbuf = Qb   + elems;
  unsigned short* Vt   = Kbuf + elems;
  unsigned short* Ob   = Vt   + elems;                // base: 64 MB

  const size_t need_base = 4 * elems * sizeof(unsigned short);
  const size_t need_full = need_base + 2 * elems * sizeof(unsigned short)
                                     + 4 * welem * sizeof(unsigned short);

  const dim3 gg(Dn / 128, Mn / 128);                  // (8, 64)
  const dim3 ga(Ln / 128, Bn, Hn);                    // (16, 4, 8)

  if (ws_size >= need_full) {
    // Pre-convert activations and weights to bf16 once: the GEMM hot loops then
    // issue pure global_load_b128 + v_wmma with no per-tile conversion VALU.
    unsigned short* Xq  = Ob  + elems;
    unsigned short* Xk  = Xq  + elems;
    unsigned short* Wqb = Xk  + elems;
    unsigned short* Wkb = Wqb + welem;
    unsigned short* Wvb = Wkb + welem;
    unsigned short* Wob = Wvb + welem;

    cvt_f32_bf16<<<(unsigned)(elems / (256 * 8)), 256, 0, stream>>>(q,  Xq,  (long)elems);
    cvt_f32_bf16<<<(unsigned)(elems / (256 * 8)), 256, 0, stream>>>(k,  Xk,  (long)elems);
    cvt_f32_bf16<<<(unsigned)(welem / (256 * 8)), 256, 0, stream>>>(Wq, Wqb, (long)welem);
    cvt_f32_bf16<<<(unsigned)(welem / (256 * 8)), 256, 0, stream>>>(Wk, Wkb, (long)welem);
    cvt_f32_bf16<<<(unsigned)(welem / (256 * 8)), 256, 0, stream>>>(Wv, Wvb, (long)welem);
    cvt_f32_bf16<<<(unsigned)(welem / (256 * 8)), 256, 0, stream>>>(Wo, Wob, (long)welem);

    gemm_xwT<unsigned short, unsigned short, unsigned short>
        <<<gg, 128, 0, stream>>>(Xq, Wqb, Qb,   Dn, (long)Dn, 1L);
    gemm_xwT<unsigned short, unsigned short, unsigned short>
        <<<gg, 128, 0, stream>>>(Xk, Wkb, Kbuf, Dn, (long)Dn, 1L);
    gemm_xwT<unsigned short, unsigned short, unsigned short>
        <<<gg, 128, 0, stream>>>(Xk, Wvb, Vt,   Dn, 1L, (long)Mn);     // V^T store

    attn_fused<<<ga, 128, 0, stream>>>(Qb, Kbuf, Vt, sl, Ob);

    gemm_xwT<unsigned short, unsigned short, float>
        <<<gg, 128, 0, stream>>>(Ob, Wob, out, Dn, (long)Dn, 1L);
  } else {
    // Fallback: convert fp32 operands on the fly inside the GEMMs.
    gemm_xwT<float, float, unsigned short>
        <<<gg, 128, 0, stream>>>(q, Wq, Qb,   Dn, (long)Dn, 1L);
    gemm_xwT<float, float, unsigned short>
        <<<gg, 128, 0, stream>>>(k, Wk, Kbuf, Dn, (long)Dn, 1L);
    gemm_xwT<float, float, unsigned short>
        <<<gg, 128, 0, stream>>>(k, Wv, Vt,   Dn, 1L, (long)Mn);

    attn_fused<<<ga, 128, 0, stream>>>(Qb, Kbuf, Vt, sl, Ob);

    gemm_xwT<unsigned short, float, float>
        <<<gg, 128, 0, stream>>>(Ob, Wo, out, Dn, (long)Dn, 1L);
  }
}
